// MultiHeadCrossAttention_24472723653249
// MI455X (gfx1250) — matrix operand order repaired and checked by reading, not compiled, not run on hardware
//
#include <hip/hip_runtime.h>
#include <hip/hip_bf16.h>

// Problem constants (match reference)
#define Bn   2
#define TQn  2048
#define TKVn 2048
#define Cn   1024
#define Hn   16
#define HDn  64

typedef __attribute__((ext_vector_type(16))) __bf16 v16bf;
typedef __attribute__((ext_vector_type(8)))  float  v8f;
typedef __attribute__((ext_vector_type(8)))  __bf16 bf16x8;
typedef __attribute__((ext_vector_type(4)))  int    v4i;
typedef __attribute__((address_space(3)))    v4i    v4i_lds;

// ---------------------------------------------------------------------------
// Async global->LDS copy (GLOBAL_LOAD_ASYNC_TO_LDS_B128, ASYNCcnt-tracked).
#if defined(__has_builtin)
#if __has_builtin(__builtin_amdgcn_global_load_async_to_lds_b128)
#define MHCA_HAS_ASYNC 1
#endif
#endif
#ifndef MHCA_HAS_ASYNC
#define MHCA_HAS_ASYNC 0
#endif

__device__ __forceinline__ void cp_b128_to_lds(const __bf16* gsrc,
                                               __bf16* ldst) {
#if MHCA_HAS_ASYNC
  __builtin_amdgcn_global_load_async_to_lds_b128(
      (v4i*)const_cast<__bf16*>(gsrc), (v4i_lds*)(v4i*)ldst, 0, 0);
#else
  *reinterpret_cast<bf16x8*>(ldst) = *reinterpret_cast<const bf16x8*>(gsrc);
#endif
}

// Wait until at most `pending` of this wave's async copies remain in flight.
// Async loads complete in order, so waiting to N leaves only the newest N.
__device__ __forceinline__ void cp_lds_wait_n(int pending) {
#if MHCA_HAS_ASYNC
  if (pending >= 4)
    asm volatile("s_wait_asynccnt 0x4" ::: "memory");
  else if (pending >= 1)
    asm volatile("s_wait_asynccnt 0x1" ::: "memory");
  else
    asm volatile("s_wait_asynccnt 0x0" ::: "memory");
#else
  (void)pending;
#endif
}

// ---------------------------------------------------------------------------
// WMMA fragment loaders (CDNA5 16-bit layouts, CDNA5 ISA §7.12.2)
//
// A matrix (16x32 bf16 tile), row-major source, leading dim ld:
//   lane L: half g = L>>4, row m = L&15
//   elems 0..7  -> K = 8g + e ; elems 8..15 -> K = 16 + 8g + (e-8)
__device__ __forceinline__ v16bf load_a_frag(const __bf16* base, int ld) {
  const int lane = threadIdx.x & 31;
  const int g = lane >> 4, m = lane & 15;
  const __bf16* p = base + (size_t)m * ld;
  bf16x8 lo = *reinterpret_cast<const bf16x8*>(p + 8 * g);
  bf16x8 hi = *reinterpret_cast<const bf16x8*>(p + 16 + 8 * g);
  v16bf f;
#pragma unroll
  for (int i = 0; i < 8; ++i) { f[i] = lo[i]; f[i + 8] = hi[i]; }
  return f;
}

// B matrix (32x16 tile). Physical source is "N-major": row-major matrix whose
// ROWS are the WMMA N dim and COLS are the contraction K.
//   lane L: half g = L>>4, col n = L&15 ; elems 0..7 -> K = 8g + e ; elems 8..15 -> K = 16 + 8g + (e-8)
__device__ __forceinline__ v16bf load_b_frag(const __bf16* base, int ld) {
  const int lane = threadIdx.x & 31;
  const int g = lane >> 4, n = lane & 15;
  const __bf16* p = base + (size_t)n * ld;
  bf16x8 lo = *reinterpret_cast<const bf16x8*>(p + 8 * g);
  bf16x8 hi = *reinterpret_cast<const bf16x8*>(p + 16 + 8 * g);
  v16bf f;
#pragma unroll
  for (int i = 0; i < 8; ++i) { f[i] = lo[i]; f[i + 8] = hi[i]; }
  return f;
}

// ---------------------------------------------------------------------------
// fp32 -> bf16 conversion (grid-stride)
__global__ void mhca_cvt_bf16(const float* __restrict__ in,
                              __bf16* __restrict__ out, int n) {
  int i = blockIdx.x * blockDim.x + threadIdx.x;
  const int stride = gridDim.x * blockDim.x;
  for (; i < n; i += stride) out[i] = (__bf16)in[i];
}

// ---------------------------------------------------------------------------
// GEMM: C[M,N] = A[M,K] · W[N,K]ᵀ   (A row-major, W row-major N x K)
// Block = 256 threads (8 waves); tile 128(M) x 64(N); K-step 32.
// The 64x32 W-tile (4 KB) is DOUBLE-BUFFERED in LDS via async global->LDS:
// tile i+1 streams in while WMMAs consume tile i.
// MODE 0: bf16 C out. MODE 1: split K rows / transposed-V store. MODE 2: f32.
template <int MODE>
__global__ __launch_bounds__(256) void mhca_gemm_wmma(
    const __bf16* __restrict__ A, const __bf16* __restrict__ Wt,
    __bf16* __restrict__ outb, __bf16* __restrict__ outb2,
    float* __restrict__ outf, int M, int N, int K) {
  __shared__ __bf16 ldsB[2][64 * 32];  // 2 x 4 KB ping-pong

  const int wave = threadIdx.x >> 5;
  const int lane = threadIdx.x & 31;
  const int row0 = blockIdx.x * 128 + wave * 16;
  const int col0 = blockIdx.y * 64;

  // staging map: thread -> one 16-byte chunk of the 64x32 tile (1 copy/thread)
  const int srow = threadIdx.x >> 2;        // 0..63  (N row)
  const int skoff = (threadIdx.x & 3) * 8;  // 0,8,16,24 (K elems)
  const __bf16* wsrc = Wt + (size_t)(col0 + srow) * K + skoff;

  const int steps = K >> 5;  // K / 32
  // prologue: stage tile 0 into buffer 0
  cp_b128_to_lds(wsrc, &ldsB[0][srow * 32 + skoff]);

  v8f acc[4] = {};
  for (int i = 0; i < steps; ++i) {
    const int cur = i & 1;
    const bool more = (i + 1) < steps;
    if (more)  // stage tile i+1 into the other buffer (overlaps compute)
      cp_b128_to_lds(wsrc + (i + 1) * 32, &ldsB[1 - cur][srow * 32 + skoff]);
    __builtin_prefetch(A + (size_t)row0 * K + i * 32 + 64, 0, 0);
    cp_lds_wait_n(more ? 1 : 0);  // this wave's tile-i copy done
    __syncthreads();              // all waves' tile-i copies done

    v16bf a = load_a_frag(A + (size_t)row0 * K + i * 32, K);
#pragma unroll
    for (int t = 0; t < 4; ++t) {
      v16bf b = load_b_frag(&ldsB[cur][(t * 16) * 32], 32);
      acc[t] = __builtin_amdgcn_wmma_f32_16x16x32_bf16(
          false, a, false, b, (short)0, acc[t], false, false);
    }
    __syncthreads();  // all waves done reading buf cur before it is restaged
  }

  // D layout: lane L holds col n=L&15; vgpr r holds row r + 8*(L>>4)
  const int g = lane >> 4, n = lane & 15;
#pragma unroll
  for (int t = 0; t < 4; ++t) {
#pragma unroll
    for (int r = 0; r < 8; ++r) {
      const int row = row0 + r + 8 * g;
      const int col = col0 + t * 16 + n;
      const float v = acc[t][r];
      if (MODE == 0) {
        outb[(size_t)row * N + col] = (__bf16)v;
      } else if (MODE == 2) {
        outf[(size_t)row * N + col] = v;
      } else {  // MODE 1: split K / transposed V
        if (col < Cn) {
          outb[(size_t)row * Cn + col] = (__bf16)v;
        } else {
          const int bb = row >> 11;          // row / TKV
          const int rl = row & (TKVn - 1);   // row % TKV
          outb2[((size_t)bb * Cn + (col - Cn)) * TKVn + rl] = (__bf16)v;
        }
      }
    }
  }
}

// ---------------------------------------------------------------------------
// Fused flash attention. 8 waves/block share one (b,h); K/V tiles (64x64
// each) are DOUBLE-BUFFERED in LDS via async global->LDS (4 copies/thread/
// step -> wait asynccnt<=4 keeps the next tile in flight during compute).
//   S (16x64)  = 8x v_wmma ;  online softmax (shfl_xor row reductions)
//   P -> per-wave LDS (D->A layout) ;  O (16x64) += P·V = 8x v_wmma
__global__ __launch_bounds__(256) void mhca_attn_wmma(
    const __bf16* __restrict__ Qb, const __bf16* __restrict__ Kb,
    const __bf16* __restrict__ Vt, const unsigned char* __restrict__ qmask,
    const unsigned char* __restrict__ kvmask, __bf16* __restrict__ Yb) {
  __shared__ __bf16 ldsK[2][64 * 64];   // 2 x 8 KB: rows = kv, cols = d
  __shared__ __bf16 ldsV[2][64 * 64];   // 2 x 8 KB: rows = d,  cols = kv
  __shared__ __bf16 ldsP[8][16 * 64];   // 2 KB per wave

  const int wave = threadIdx.x >> 5;
  const int lane = threadIdx.x & 31;
  const int g = lane >> 4, n = lane & 15;

  const int tile = blockIdx.x * 8 + wave;  // 0 .. B*H*(TQ/16)-1
  const int qt = tile & 127;               // TQ/16 = 128
  const int h = (tile >> 7) & (Hn - 1);    // same for all 8 waves in block
  const int b = tile >> 11;
  const int q0 = qt * 16;

  const __bf16* Qh = Qb + ((size_t)(b * TQn + q0)) * Cn + h * HDn;
  const __bf16* Kh = Kb + (size_t)b * TKVn * Cn + h * HDn;
  const __bf16* Vh = Vt + ((size_t)b * Cn + h * HDn) * TKVn;

  // staging map: thread -> one row, 16 elems (two b128 chunks per matrix)
  const int sr = threadIdx.x >> 2;        // 0..63
  const int so = (threadIdx.x & 3) * 16;  // 0,16,32,48

  // Q fragments, pre-scaled by 1/sqrt(HD) = 0.125 (exact in bf16)
  v16bf qf[2];
#pragma unroll
  for (int kc = 0; kc < 2; ++kc) {
    v16bf tmp = load_a_frag(Qh + kc * 32, Cn);
#pragma unroll
    for (int i = 0; i < 16; ++i) tmp[i] = (__bf16)((float)tmp[i] * 0.125f);
    qf[kc] = tmp;
  }

  bool qm[8];
#pragma unroll
  for (int r = 0; r < 8; ++r)
    qm[r] = qmask[b * TQn + q0 + r + 8 * g] != 0;

  float runmax[8], runsum[8];
  v8f o[4] = {};
#pragma unroll
  for (int r = 0; r < 8; ++r) { runmax[r] = -3.0e38f; runsum[r] = 0.f; }

  // prologue: stage kv block 0 into buffer 0
  {
    const __bf16* gk = Kh + (size_t)sr * Cn + so;
    cp_b128_to_lds(gk, &ldsK[0][sr * 64 + so]);
    cp_b128_to_lds(gk + 8, &ldsK[0][sr * 64 + so + 8]);
    const __bf16* gv = Vh + (size_t)sr * TKVn + so;
    cp_b128_to_lds(gv, &ldsV[0][sr * 64 + so]);
    cp_b128_to_lds(gv + 8, &ldsV[0][sr * 64 + so + 8]);
  }

  for (int kv0 = 0; kv0 < TKVn; kv0 += 64) {
    const int cur = (kv0 >> 6) & 1;
    const bool more = (kv0 + 64) < TKVn;
    if (more) {  // stage next kv block into the other buffer
      const int nxt = 1 - cur;
      const __bf16* gk = Kh + (size_t)(kv0 + 64 + sr) * Cn + so;
      cp_b128_to_lds(gk, &ldsK[nxt][sr * 64 + so]);
      cp_b128_to_lds(gk + 8, &ldsK[nxt][sr * 64 + so + 8]);
      const __bf16* gv = Vh + (size_t)sr * TKVn + kv0 + 64 + so;
      cp_b128_to_lds(gv, &ldsV[nxt][sr * 64 + so]);
      cp_b128_to_lds(gv + 8, &ldsV[nxt][sr * 64 + so + 8]);
    }
    cp_lds_wait_n(more ? 4 : 0);  // this wave's current-tile copies done
    __syncthreads();              // all waves' current-tile copies done

    const __bf16* bK = &ldsK[cur][0];
    const __bf16* bV = &ldsV[cur][0];

    // ---- scores S = (Q/8) · Kᵀ for 4 N-tiles of 16 kv each
    float s[4][8];
#pragma unroll
    for (int t = 0; t < 4; ++t) {
      v8f sc = {};
      v16bf kf0 = load_b_frag(bK + (t * 16) * 64 + 0, 64);
      sc = __builtin_amdgcn_wmma_f32_16x16x32_bf16(false, qf[0], false, kf0,
                                                   (short)0, sc, false, false);
      v16bf kf1 = load_b_frag(bK + (t * 16) * 64 + 32, 64);
      sc = __builtin_amdgcn_wmma_f32_16x16x32_bf16(false, qf[1], false, kf1,
                                                   (short)0, sc, false, false);
      const bool km = kvmask[b * TKVn + kv0 + t * 16 + n] != 0;
#pragma unroll
      for (int r = 0; r < 8; ++r)
        s[t][r] = (km && qm[r]) ? sc[r] : -1.0e9f;
    }

    // ---- online softmax: row reductions across the 16-lane half
    float nm[8], scl[8], rs[8];
#pragma unroll
    for (int r = 0; r < 8; ++r) {
      float m = fmaxf(fmaxf(s[0][r], s[1][r]), fmaxf(s[2][r], s[3][r]));
      m = fmaxf(m, __shfl_xor(m, 1, 32));
      m = fmaxf(m, __shfl_xor(m, 2, 32));
      m = fmaxf(m, __shfl_xor(m, 4, 32));
      m = fmaxf(m, __shfl_xor(m, 8, 32));
      nm[r] = fmaxf(runmax[r], m);
      scl[r] = __expf(runmax[r] - nm[r]);
      runmax[r] = nm[r];
      rs[r] = 0.f;
    }
#pragma unroll
    for (int t = 0; t < 4; ++t)
#pragma unroll
      for (int r = 0; r < 8; ++r) {
        const float p = __expf(s[t][r] - nm[r]);
        s[t][r] = p;
        rs[r] += p;
      }
#pragma unroll
    for (int r = 0; r < 8; ++r) {
      float x = rs[r];
      x += __shfl_xor(x, 1, 32);
      x += __shfl_xor(x, 2, 32);
      x += __shfl_xor(x, 4, 32);
      x += __shfl_xor(x, 8, 32);
      runsum[r] = runsum[r] * scl[r] + x;
    }
#pragma unroll
    for (int t = 0; t < 4; ++t)
#pragma unroll
      for (int r = 0; r < 8; ++r) o[t][r] *= scl[r];

    // ---- P (D layout) -> per-wave LDS -> A fragments
    __bf16* lp = &ldsP[wave][0];
#pragma unroll
    for (int t = 0; t < 4; ++t)
#pragma unroll
      for (int r = 0; r < 8; ++r)
        lp[(r + 8 * g) * 64 + t * 16 + n] = (__bf16)s[t][r];
    asm volatile("s_wait_dscnt 0x0" ::: "memory");  // LDS RAW within wave
    v16bf pa0 = load_a_frag(lp + 0, 64);
    v16bf pa1 = load_a_frag(lp + 32, 64);

    // ---- O += P · V  (V B-frags from LDS, d-major rows)
#pragma unroll
    for (int t = 0; t < 4; ++t) {
      v16bf v0 = load_b_frag(bV + (t * 16) * 64 + 0, 64);
      v16bf v1 = load_b_frag(bV + (t * 16) * 64 + 32, 64);
      o[t] = __builtin_amdgcn_wmma_f32_16x16x32_bf16(false, pa0, false, v0,
                                                     (short)0, o[t], false, false);
      o[t] = __builtin_amdgcn_wmma_f32_16x16x32_bf16(false, pa1, false, v1,
                                                     (short)0, o[t], false, false);
    }
    __syncthreads();  // all waves done reading buf cur before it is restaged
  }

  // ---- normalize and store Y (bf16, natural (B,TQ,C) layout)
#pragma unroll
  for (int r = 0; r < 8; ++r) {
    const float inv = 1.0f / runsum[r];
    const size_t row = (size_t)(b * TQn + q0 + r + 8 * g);
#pragma unroll
    for (int t = 0; t < 4; ++t)
      Yb[row * Cn + h * HDn + t * 16 + n] = (__bf16)(o[t][r] * inv);
  }
}

// ---------------------------------------------------------------------------
extern "C" void kernel_launch(void* const* d_in, const int* in_sizes, int n_in,
                              void* d_out, int out_size, void* d_ws,
                              size_t ws_size, hipStream_t stream) {
  (void)in_sizes; (void)n_in; (void)out_size; (void)ws_size;

  const float* x_q  = (const float*)d_in[0];
  const float* x_kv = (const float*)d_in[1];
  const unsigned char* qm = (const unsigned char*)d_in[2];
  const unsigned char* km = (const unsigned char*)d_in[3];
  const float* Wq   = (const float*)d_in[4];
  const float* Wkv  = (const float*)d_in[5];
  const float* Wc   = (const float*)d_in[6];
  float* out = (float*)d_out;

  // Workspace layout (bf16 elements), ~56 MB total
  __bf16* ws = (__bf16*)d_ws;
  size_t off = 0;
  const size_t MT = (size_t)Bn * TQn;  // 4096 rows
  __bf16* Xq   = ws + off; off += MT * Cn;
  __bf16* Xkv  = ws + off; off += MT * Cn;
  __bf16* Wqb  = ws + off; off += (size_t)Cn * Cn;
  __bf16* Wkvb = ws + off; off += (size_t)2 * Cn * Cn;
  __bf16* Wcb  = ws + off; off += (size_t)Cn * Cn;
  __bf16* Qb   = ws + off; off += MT * Cn;
  __bf16* Kbf  = ws + off; off += MT * Cn;
  __bf16* Vtb  = ws + off; off += MT * Cn;
  __bf16* Yb   = ws + off; off += MT * Cn;

  // 1) fp32 -> bf16
  mhca_cvt_bf16<<<2048, 256, 0, stream>>>(x_q,  Xq,  (int)(MT * Cn));
  mhca_cvt_bf16<<<2048, 256, 0, stream>>>(x_kv, Xkv, (int)(MT * Cn));
  mhca_cvt_bf16<<<512,  256, 0, stream>>>(Wq,   Wqb,  Cn * Cn);
  mhca_cvt_bf16<<<1024, 256, 0, stream>>>(Wkv,  Wkvb, 2 * Cn * Cn);
  mhca_cvt_bf16<<<512,  256, 0, stream>>>(Wc,   Wcb,  Cn * Cn);

  // 2) projections
  mhca_gemm_wmma<0><<<dim3(32, 16), 256, 0, stream>>>(
      Xq, Wqb, Qb, nullptr, nullptr, 4096, Cn, Cn);
  mhca_gemm_wmma<1><<<dim3(32, 32), 256, 0, stream>>>(
      Xkv, Wkvb, Kbf, Vtb, nullptr, 4096, 2 * Cn, Cn);

  // 3) fused flash attention: B*H*(TQ/16)/8 = 512 blocks
  mhca_attn_wmma<<<512, 256, 0, stream>>>(Qb, Kbf, Vtb, qm, km, Yb);

  // 4) output projection -> fp32 d_out
  mhca_gemm_wmma<2><<<dim3(32, 16), 256, 0, stream>>>(
      Yb, Wcb, nullptr, nullptr, out, 4096, Cn, Cn);
}
